// CausalSelfAttention_88046829568450
// MI455X (gfx1250) — matrix operand order repaired and checked by reading, not compiled, not run on hardware
//
#include <hip/hip_runtime.h>
#include <hip/hip_bf16.h>
#include <stdint.h>

// Problem constants (match reference)
constexpr int kB  = 2;
constexpr int kT  = 2048;
constexpr int kC  = 1024;
constexpr int kH  = 16;
constexpr int kHD = 64;
constexpr int kC3 = 3 * kC;
constexpr int kJ  = kHD / 2;   // rotary pairs per head

typedef __attribute__((ext_vector_type(16))) __bf16 v16bf;
typedef __attribute__((ext_vector_type(8)))  float  v8f;

// ---------------- helpers ----------------
__device__ __forceinline__ unsigned short f2bf(float f) {
    unsigned int u = __builtin_bit_cast(unsigned int, f);
    unsigned int r = u + 0x7FFFu + ((u >> 16) & 1u);   // round-to-nearest-even
    return (unsigned short)(r >> 16);
}
__device__ __forceinline__ float bf2f(unsigned short h) {
    unsigned int u = ((unsigned int)h) << 16;
    return __builtin_bit_cast(float, u);
}

__device__ __forceinline__ v8f wmma_bf16(v16bf a, v16bf b, v8f c) {
    // D = A(16x32 bf16) * B(32x16 bf16) + C(16x16 f32)
    return __builtin_amdgcn_wmma_f32_16x16x32_bf16(
        false, a, false, b, (short)0, c, false, false);
}

// Async copy of one 16-byte chunk: global -> LDS, tracked by ASYNCcnt.
// GV addressing: vdst VGPR = wave-relative LDS byte offset (= low 32 bits of
// the generic pointer), vaddr = 64-bit global address.
__device__ __forceinline__ void async_copy16(void* ldsDst, const void* gSrc) {
    uint32_t ldsOff = (uint32_t)(uintptr_t)ldsDst;
    uint64_t gaddr  = (uint64_t)(uintptr_t)gSrc;
    asm volatile("global_load_async_to_lds_b128 %0, %1, off"
                 :: "v"(ldsOff), "v"(gaddr) : "memory");
}
__device__ __forceinline__ void async_wait0() {
    asm volatile("s_wait_asynccnt 0x0" ::: "memory");
}

// A-fragment (16x32, bf16) from row-major A[*][lda], tile rows mbase..mbase+15,
// contraction cols kb..kb+31.  ISA layout: lanes 0-15 hold K 0-7 (elems 0-7) and
// K 16-23 (elems 8-15); lanes 16-31 hold K 8-15 and K 24-31.
__device__ __forceinline__ v16bf load_a_frag(const unsigned short* A,
                                             int lda, int mbase, int kb) {
    int lane = threadIdx.x & 31;
    int row  = mbase + (lane & 15);
    int koff = kb + ((lane >> 4) << 3);          // +8 for upper half-wave
    union { uint4 u[2]; v16bf v; } r;
    r.u[0] = *(const uint4*)(A + (size_t)row * lda + koff);
    r.u[1] = *(const uint4*)(A + (size_t)row * lda + koff + 16);
    return r.v;
}

// B-fragment (32x16, bf16) from Bt stored N-major: Bt[n][ldb] holds column n of B
// with contraction dim contiguous.  Lanes 0-15: K kb..kb+7 and kb+16..kb+23 for col n=lane;
// lanes 16-31: K kb+8..kb+15 and kb+24..kb+31.
__device__ __forceinline__ v16bf load_b_frag(const unsigned short* Bt,
                                             int ldb, int nbase, int kb) {
    int lane = threadIdx.x & 31;
    int col  = nbase + (lane & 15);
    int koff = kb + ((lane >> 4) << 3);          // +8 for upper half-wave
    union { uint4 u[2]; v16bf v; } r;
    const uint4* p = (const uint4*)(Bt + (size_t)col * ldb + koff);
    r.u[0] = p[0];
    r.u[1] = p[2];
    return r.v;
}

// ---------------- cast kernels ----------------
__global__ void cast_f32_bf16(const float* __restrict__ in,
                              unsigned short* __restrict__ out, int n) {
    int i = blockIdx.x * blockDim.x + threadIdx.x;
    if (i < n) out[i] = f2bf(in[i]);
}

// in: [R][N] row-major fp32  ->  out: [N][R] row-major bf16 (transposed)
__global__ void cast_transpose_bf16(const float* __restrict__ in,
                                    unsigned short* __restrict__ out,
                                    int R, int N) {
    int i = blockIdx.x * blockDim.x + threadIdx.x;
    if (i >= R * N) return;
    int k = i / N, n = i % N;
    out[(size_t)n * R + k] = f2bf(in[i]);
}

// ---------------- LDS-tiled, double-buffered WMMA GEMM ----------------
//   Out[M][N] = A[M][K](bf16) * Bt[N][K]^T(bf16) + bias
// Block: 256 threads = 8 waves, 128(M) x 64(N) tile; wave = 32x32 (2x2 WMMA).
// K-step 32.  A/B tiles staged in LDS via async global->LDS loads (ASYNCcnt),
// rows padded to 40 elements (20 dwords) -> conflict-free ds_load_b128.
constexpr int kLdPad = 40;                         // 32 + 8 bf16 elements
constexpr int kABufE = 128 * kLdPad;               // 5120 elems / buffer
constexpr int kBBufE = 64 * kLdPad;                // 2560 elems / buffer
constexpr int kBufE  = kABufE + kBBufE;            // 7680 elems / buffer

template <bool OUTF32>
__global__ void gemm_wmma_tiled(const unsigned short* __restrict__ A,
                                const unsigned short* __restrict__ Bt,
                                const float* __restrict__ bias,
                                void* __restrict__ Out,
                                int M, int N, int K) {
    __shared__ __align__(16) unsigned short smem[2 * kBufE];

    const int tid   = threadIdx.x;
    const int wid   = __builtin_amdgcn_readfirstlane(tid >> 5);
    const int lane  = tid & 31;
    const int nbase = blockIdx.x * 64;             // gridDim.x = N/64
    const int mbase = blockIdx.y * 128;            // gridDim.y = M/128
    const int wm    = (wid >> 1) * 32;             // wave row offset in tile
    const int wn    = (wid & 1) * 32;              // wave col offset in tile

    auto issue_loads = [&](int buf, int kb) {
        unsigned short* sA = smem + buf * kBufE;
        unsigned short* sB = sA + kABufE;
        // A tile: 128 rows x 32 K = 512 x 16B chunks, 2 per thread
        for (int c = tid; c < 512; c += 256) {
            int row = c >> 2, quad = c & 3;
            async_copy16(sA + row * kLdPad + quad * 8,
                         A + (size_t)(mbase + row) * K + kb + quad * 8);
        }
        // B tile: 64 cols x 32 K = 256 x 16B chunks, 1 per thread
        {
            int col = tid >> 2, quad = tid & 3;
            async_copy16(sB + col * kLdPad + quad * 8,
                         Bt + (size_t)(nbase + col) * K + kb + quad * 8);
        }
    };

    v8f acc[2][2];
    for (int mi = 0; mi < 2; ++mi)
        for (int ni = 0; ni < 2; ++ni)
            for (int i = 0; i < 8; ++i) acc[mi][ni][i] = 0.f;

    issue_loads(0, 0);
    int nsteps = K >> 5;
    for (int it = 0; it < nsteps; ++it) {
        int cur = it & 1;
        async_wait0();          // my async loads into `cur` complete
        __syncthreads();        // everyone's loads done; everyone done reading nxt
        if (it + 1 < nsteps) issue_loads(cur ^ 1, (it + 1) << 5);

        const unsigned short* sA = smem + cur * kBufE;
        const unsigned short* sB = sA + kABufE;
        v16bf a0 = load_a_frag(sA, kLdPad, wm,      0);
        v16bf a1 = load_a_frag(sA, kLdPad, wm + 16, 0);
        v16bf b0 = load_b_frag(sB, kLdPad, wn,      0);
        v16bf b1 = load_b_frag(sB, kLdPad, wn + 16, 0);
        acc[0][0] = wmma_bf16(a0, b0, acc[0][0]);
        acc[0][1] = wmma_bf16(a0, b1, acc[0][1]);
        acc[1][0] = wmma_bf16(a1, b0, acc[1][0]);
        acc[1][1] = wmma_bf16(a1, b1, acc[1][1]);
    }

    // Epilogue: C layout -> row = v + 8*(lane/16), col = lane&15
    for (int mi = 0; mi < 2; ++mi) {
        int rbase = mbase + wm + mi * 16 + ((lane >> 4) << 3);
        for (int ni = 0; ni < 2; ++ni) {
            int n  = nbase + wn + ni * 16 + (lane & 15);
            float bv = bias ? bias[n] : 0.f;
            if (OUTF32) {
                float* O = (float*)Out;
                for (int v = 0; v < 8; ++v)
                    O[(size_t)(rbase + v) * N + n] = acc[mi][ni][v] + bv;
            } else {
                unsigned short* O = (unsigned short*)Out;
                for (int v = 0; v < 8; ++v)
                    O[(size_t)(rbase + v) * N + n] = f2bf(acc[mi][ni][v] + bv);
            }
        }
    }
}

// ---------------- RoPE + head packing ----------------
// qkvb: bf16 [B*T][3C].  Writes:
//   Qb [bh][t][d]  (rotary, pre-scaled by 1/sqrt(HD))
//   Kb [bh][t][d]  (rotary)        -> B-fragment layout for S = Q K^T
//   Vt [bh][d][t]  (transposed)    -> B-fragment layout for O = P V
__global__ void rope_prep(const unsigned short* __restrict__ qkvb,
                          const float* __restrict__ cosT,
                          const float* __restrict__ sinT,
                          unsigned short* __restrict__ Qb,
                          unsigned short* __restrict__ Kb,
                          unsigned short* __restrict__ Vt) {
    int i = blockIdx.x * blockDim.x + threadIdx.x;
    if (i >= kB * kT * kH * kJ) return;
    int j   = i % kJ;  int tmp = i / kJ;
    int h   = tmp % kH; tmp /= kH;
    int t   = tmp % kT;
    int bi  = tmp / kT;

    size_t base = ((size_t)(bi * kT + t)) * kC3 + h * kHD + 2 * j;
    float q0 = bf2f(qkvb[base]),          q1 = bf2f(qkvb[base + 1]);
    float k0 = bf2f(qkvb[base + kC]),     k1 = bf2f(qkvb[base + kC + 1]);
    float v0 = bf2f(qkvb[base + 2 * kC]), v1 = bf2f(qkvb[base + 2 * kC + 1]);
    float cs = cosT[t * kJ + j], sn = sinT[t * kJ + j];

    float qe = q0 * cs - q1 * sn, qo = q0 * sn + q1 * cs;
    float ke = k0 * cs - k1 * sn, ko = k0 * sn + k1 * cs;
    const float scale = 0.125f;                  // 1/sqrt(64)

    size_t qk = ((size_t)((bi * kH + h) * kT + t)) * kHD + 2 * j;
    Qb[qk]     = f2bf(qe * scale);
    Qb[qk + 1] = f2bf(qo * scale);
    Kb[qk]     = f2bf(ke);
    Kb[qk + 1] = f2bf(ko);

    size_t vb = ((size_t)(bi * kH + h) * kHD + 2 * j) * kT + t;
    Vt[vb]      = f2bf(v0);
    Vt[vb + kT] = f2bf(v1);
}

// ---------------- flash attention (4 waves per block, 1 q-tile each) ----------------
__global__ void attn_fa(const unsigned short* __restrict__ Qb,
                        const unsigned short* __restrict__ Kb,
                        const unsigned short* __restrict__ Vt,
                        unsigned short* __restrict__ Y) {
    __shared__ unsigned short pT[4][16 * 32];    // per-wave P staging (C->A layout)

    int bh = blockIdx.x;                         // 0..B*H-1
    int wv = __builtin_amdgcn_readfirstlane(threadIdx.x >> 5);   // wave-uniform
    int qt = blockIdx.y * 4 + wv;                // 0..T/16-1 (scalar)
    int bi = bh / kH, h = bh % kH;
    int lane  = threadIdx.x & 31;
    int n     = lane & 15;
    int rbase = (lane >> 4) << 3;
    unsigned short* myP = pT[wv];

    const unsigned short* Qh = Qb + (size_t)bh * kT * kHD;
    const unsigned short* Kh = Kb + (size_t)bh * kT * kHD;
    const unsigned short* Vh = Vt + (size_t)bh * kHD * kT;

    int qbase = qt * 16;

    // Q tile (16 x 64) as two A-fragments, reused across all key blocks.
    v16bf qf0 = load_a_frag(Qh, kHD, qbase, 0);
    v16bf qf1 = load_a_frag(Qh, kHD, qbase, 32);

    v8f o[4];
    for (int d = 0; d < 4; ++d)
        for (int i = 0; i < 8; ++i) o[d][i] = 0.f;
    float mrow[8], lrow[8];
    for (int v = 0; v < 8; ++v) { mrow[v] = -1e30f; lrow[v] = 0.f; }

    const int kend = qbase + 16;                 // causal: keys < kend (scalar)
    for (int kb0 = 0; kb0 < kend; kb0 += 32) {
        // ---- issue ALL global loads for this block up front:
        //      K fragments (S-tiles) and V fragments (PV) in one clause,
        //      so they drain under the softmax VALU work.
        const bool tile1 = (kb0 + 16) < kend;    // scalar branch
        v16bf kf00 = load_b_frag(Kh, kHD, kb0, 0);
        v16bf kf01 = load_b_frag(Kh, kHD, kb0, 32);
        v16bf kf10 = kf00, kf11 = kf01;
        if (tile1) {
            kf10 = load_b_frag(Kh, kHD, kb0 + 16, 0);
            kf11 = load_b_frag(Kh, kHD, kb0 + 16, 32);
        }
        v16bf vf0 = load_b_frag(Vh, kT, 0,  kb0);
        v16bf vf1 = load_b_frag(Vh, kT, 16, kb0);
        v16bf vf2 = load_b_frag(Vh, kT, 32, kb0);
        v16bf vf3 = load_b_frag(Vh, kT, 48, kb0);

        // ---- S = (Q/sqrt(d)) K^T for the 16x32 key block (two 16x16 tiles) ----
        v8f s[2];
        {
            v8f z;
            for (int i = 0; i < 8; ++i) z[i] = 0.f;
            v8f s0 = wmma_bf16(qf0, kf00, z);
            s0     = wmma_bf16(qf1, kf01, s0);
            s[0] = s0;
            if (tile1) {
                v8f s1 = wmma_bf16(qf0, kf10, z);
                s1     = wmma_bf16(qf1, kf11, s1);
                s[1] = s1;
            } else {
                s[1] = z;
            }
        }

        // ---- causal mask + per-row partial max ----
        float pmax[8];
        for (int v = 0; v < 8; ++v) {
            int q  = qbase + rbase + v;
            int c0 = kb0 + n, c1 = kb0 + 16 + n;
            float s0 = (c0 > q) ? -1e30f : s[0][v];
            float s1 = (c1 > q) ? -1e30f : s[1][v];
            s[0][v] = s0; s[1][v] = s1;
            pmax[v] = fmaxf(s0, s1);
        }
        for (int off = 1; off < 16; off <<= 1)
            for (int v = 0; v < 8; ++v)
                pmax[v] = fmaxf(pmax[v], __shfl_xor(pmax[v], off, 32));

        // ---- online softmax update; write P (bf16) to LDS ----
        float psum[8];
        for (int v = 0; v < 8; ++v) {
            float mn   = fmaxf(mrow[v], pmax[v]);
            float corr = __expf(mrow[v] - mn);
            float e0   = __expf(s[0][v] - mn);
            float e1   = __expf(s[1][v] - mn);
            mrow[v] = mn;
            lrow[v] *= corr;
            psum[v] = e0 + e1;
            for (int d = 0; d < 4; ++d) o[d][v] *= corr;
            myP[(rbase + v) * 32 + n]      = f2bf(e0);
            myP[(rbase + v) * 32 + 16 + n] = f2bf(e1);
        }
        for (int off = 1; off < 16; off <<= 1)
            for (int v = 0; v < 8; ++v)
                psum[v] += __shfl_xor(psum[v], off, 32);
        for (int v = 0; v < 8; ++v) lrow[v] += psum[v];

        // DS ops are in-order per wave; make the cross-lane LDS handoff explicit
        asm volatile("s_wait_dscnt 0" ::: "memory");

        // ---- read P back as an A-fragment (16x32); O += P V (4 tiles of HD) ----
        {
            int row  = lane & 15;
            int koff = (lane >> 4) << 3;
            union { uint4 u[2]; v16bf v; } r;
            r.u[0] = *(const uint4*)&myP[row * 32 + koff];
            r.u[1] = *(const uint4*)&myP[row * 32 + koff + 16];
            v16bf pa = r.v;
            o[0] = wmma_bf16(pa, vf0, o[0]);
            o[1] = wmma_bf16(pa, vf1, o[1]);
            o[2] = wmma_bf16(pa, vf2, o[2]);
            o[3] = wmma_bf16(pa, vf3, o[3]);
        }
    }

    // ---- normalize and write Y[b][t][h*64+d] (bf16) ----
    for (int v = 0; v < 8; ++v) {
        float inv = 1.0f / lrow[v];
        int t = qbase + rbase + v;
        for (int d = 0; d < 4; ++d) {
            int dd = d * 16 + n;
            Y[((size_t)(bi * kT + t)) * kC + h * kHD + dd] = f2bf(o[d][v] * inv);
        }
    }
}

// ---------------- host launcher ----------------
extern "C" void kernel_launch(void* const* d_in, const int* in_sizes, int n_in,
                              void* d_out, int out_size, void* d_ws, size_t ws_size,
                              hipStream_t stream) {
    (void)in_sizes; (void)n_in; (void)out_size; (void)ws_size;
    const float* x      = (const float*)d_in[0];
    const float* cosT   = (const float*)d_in[1];
    const float* sinT   = (const float*)d_in[2];
    const float* w_attn = (const float*)d_in[3];
    const float* b_attn = (const float*)d_in[4];
    const float* w_proj = (const float*)d_in[5];
    const float* b_proj = (const float*)d_in[6];

    char* ws = (char*)d_ws;
    unsigned short* xb   = (unsigned short*)ws; ws += (size_t)kB * kT * kC  * 2;
    unsigned short* waT  = (unsigned short*)ws; ws += (size_t)kC3 * kC      * 2;
    unsigned short* wpT  = (unsigned short*)ws; ws += (size_t)kC  * kC      * 2;
    unsigned short* qkvb = (unsigned short*)ws; ws += (size_t)kB * kT * kC3 * 2;
    unsigned short* Qb   = (unsigned short*)ws; ws += (size_t)kB * kH * kT * kHD * 2;
    unsigned short* Kb   = (unsigned short*)ws; ws += (size_t)kB * kH * kT * kHD * 2;
    unsigned short* Vt   = (unsigned short*)ws; ws += (size_t)kB * kH * kHD * kT * 2;
    unsigned short* Yb   = (unsigned short*)ws;

    const int M = kB * kT;                       // 4096

    // 1) casts / transposes to bf16
    cast_f32_bf16<<<(M * kC + 255) / 256, 256, 0, stream>>>(x, xb, M * kC);
    cast_transpose_bf16<<<(kC * kC3 + 255) / 256, 256, 0, stream>>>(w_attn, waT, kC, kC3);
    cast_transpose_bf16<<<(kC * kC  + 255) / 256, 256, 0, stream>>>(w_proj, wpT, kC, kC);

    // 2) qkv = x @ w_attn + b_attn   (bf16 out)
    gemm_wmma_tiled<false><<<dim3(kC3 / 64, M / 128), 256, 0, stream>>>(
        xb, waT, b_attn, qkvb, M, kC3, kC);

    // 3) rotary + head packing
    {
        int nthr = kB * kT * kH * kJ;
        rope_prep<<<(nthr + 255) / 256, 256, 0, stream>>>(qkvb, cosT, sinT, Qb, Kb, Vt);
    }

    // 4) flash attention (4 q-tiles per block)
    attn_fa<<<dim3(kB * kH, kT / 64), 128, 0, stream>>>(Qb, Kb, Vt, Yb);

    // 5) out = Y @ w_proj + b_proj   (fp32 out)
    gemm_wmma_tiled<true><<<dim3(kC / 64, M / 128), 256, 0, stream>>>(
        Yb, wpT, b_proj, d_out, M, kC, kC);
}
